// MultiHeadAttention_66932770341016
// MI455X (gfx1250) — matrix operand order repaired and checked by reading, not compiled, not run on hardware
//
#include <hip/hip_runtime.h>
#include <hip/hip_bf16.h>
#include <stdint.h>

// MHA forward for MI455X (gfx1250, wave32, WMMA 16x16x32 f16->f32).
// Pipeline: 3x (GEMM+bias, f32 in -> f16 out) -> flash attention (f16) ->
// GEMM+bias (f16 in -> f32 out).  Workspace: Qp,Kp,Vp,Ctx f16, 4*16MB = 64MB.
// New this round: Tensor Data Mover (tensor_load_to_lds) for the pure-f16 2D
// tile stages (Q/K tiles in attention, A tile of the output projection), with
// hardware LDS row padding via the D# pad fields, drained by s_wait_tensorcnt.

typedef __attribute__((ext_vector_type(16))) _Float16 v16h;
typedef __attribute__((ext_vector_type(8)))  _Float16 v8h;
typedef __attribute__((ext_vector_type(8)))  float    v8f;
typedef __attribute__((ext_vector_type(4)))  float    v4f;
typedef __attribute__((ext_vector_type(4)))  unsigned int u32x4;
typedef __attribute__((ext_vector_type(8)))  int      i32x8;
typedef __attribute__((ext_vector_type(4)))  int      i32x4;

#if __has_builtin(__builtin_amdgcn_tensor_load_to_lds) && \
    __has_builtin(__builtin_amdgcn_s_wait_tensorcnt)
#define HAS_TDM 1
#else
#define HAS_TDM 0
#endif

// ---------------------------------------------------------------------------
// TDM: DMA a tile_h x tile_w (f16 elements) tile from global (row stride
// stride_elems) into LDS, inserting pad_amount DWORDs of padding after every
// 2^(pad_interval_code+1) DWORDs (i.e. after each row).  D# layout per
// CDNA5 ISA.  Issue from one wave only.
// ---------------------------------------------------------------------------
static __device__ __forceinline__ void tdm_load_tile_f16(
    _Float16* lds_dst, const _Float16* gsrc, int tile_w, int tile_h,
    int stride_elems, int pad_interval_code, int pad_amount_code) {
#if HAS_TDM
  const unsigned long long ga = (unsigned long long)(uintptr_t)gsrc;
  u32x4 g0;
  g0[0] = 1u;                                    // count=1, user descriptor
  g0[1] = (unsigned int)(uintptr_t)lds_dst;      // lds_addr (byte offset)
  g0[2] = (unsigned int)ga;                      // global_addr[31:0]
  g0[3] = (unsigned int)((ga >> 32) & 0x01FFFFFFu) | (2u << 30); // [56:32]|type=2
  i32x8 g1;
  g1[0] = (1 << 16)                              // data_size = 2 bytes
        | (1 << 20)                              // pad_enable
        | (pad_interval_code << 22)              // DWORDs before pad = 2^(c+1)
        | (pad_amount_code << 25);               // pad DWORDs = c+1
  g1[1] = (tile_w & 0xFFFF) << 16;               // tensor_dim0 [15:0]
  g1[2] = ((unsigned)tile_w >> 16) | ((tile_h & 0xFFFF) << 16); // d0 hi | d1 lo
  g1[3] = ((unsigned)tile_h >> 16) | ((tile_w & 0xFFFF) << 16); // d1 hi | tile_dim0
  g1[4] = (tile_h & 0xFFFF);                     // tile_dim1 (tile_dim2 = 0)
  g1[5] = stride_elems;                          // tensor_dim0_stride [31:0]
  g1[6] = 0;
  g1[7] = 0;
  const i32x4 gz = {0, 0, 0, 0};                 // 2D tensor: groups 2/3 zero
#if defined(__clang_major__) && (__clang_major__ >= 23)
  const i32x8 gz8 = {0, 0, 0, 0, 0, 0, 0, 0};
  __builtin_amdgcn_tensor_load_to_lds(g0, g1, gz, gz, gz8, 0);
#else
  __builtin_amdgcn_tensor_load_to_lds(g0, g1, gz, gz, 0);
#endif
#endif
}

static __device__ __forceinline__ v16h cat16(v8h lo, v8h hi) {
  return __builtin_shufflevector(lo, hi, 0,1,2,3,4,5,6,7,8,9,10,11,12,13,14,15);
}

// A fragment (16x32 f16, MxK): lane L -> row M = L%16, K-half = L/16.
// Per ISA table: VGPR0-3 = K[kg*8 .. kg*8+7], VGPR4-7 = K[kg*8+16 .. +23].
static __device__ __forceinline__ v16h load_frag_a(const _Float16* tile, int stride,
                                                   int lane, int koff) {
  const int m = lane & 15, kg = lane >> 4;
  const _Float16* p = tile + m * stride + koff + kg * 8;
  v8h lo = *(const v8h*)p;
  v8h hi = *(const v8h*)(p + 16);
  return cat16(lo, hi);
}

// B fragment (32x16 f16, KxN), tile stored [n][k] row-major:
// lane L -> n = L%16, K-group = L/16; K[kg*8..+7] and K[kg*8+16..+23].
static __device__ __forceinline__ v16h load_frag_b(const _Float16* tile, int stride,
                                                   int lane, int koff) {
  const int n = lane & 15, kg = lane >> 4;
  const _Float16* p = tile + n * stride + koff + kg * 8;
  v8h lo = *(const v8h*)p;
  v8h hi = *(const v8h*)(p + 16);
  return cat16(lo, hi);
}

static __device__ __forceinline__ v8f wmma32(v16h a, v16h b, v8f c) {
  return __builtin_amdgcn_wmma_f32_16x16x32_f16(false, a, false, b, (short)0, c,
                                                false, false);
}

static __device__ __forceinline__ v8h load8(const float* p) {
  v4f a = *(const v4f*)p;
  v4f b = *(const v4f*)(p + 4);
  v8h h;
#pragma unroll
  for (int i = 0; i < 4; ++i) { h[i] = (_Float16)a[i]; h[i + 4] = (_Float16)b[i]; }
  return h;
}
static __device__ __forceinline__ v8h load8(const _Float16* p) {
  return *(const v8h*)p;
}

static __device__ __forceinline__ void store_out(float* p, float v)    { *p = v; }
static __device__ __forceinline__ void store_out(_Float16* p, float v) { *p = (_Float16)v; }

// ---------------------------------------------------------------------------
// Y[M,N] = X[M,K] @ W[N,K]^T + bias[N]   (torch Linear)
// Block: 256 threads = 8 waves, tile BM=128 x BN=64, BK=32.
// Wave grid 4x2; each wave computes 2x2 WMMA tiles (32x32).
// f16 X path stages the A tile with the TDM; f32 paths convert while staging.
// ---------------------------------------------------------------------------
template <typename XT, typename OT>
__global__ __launch_bounds__(256, 2)
void gemm_bias_wmma(const XT* __restrict__ X, const float* __restrict__ W,
                    const float* __restrict__ bias, OT* __restrict__ Y,
                    int M, int N, int K) {
  constexpr int BM = 128, BN = 64, BK = 32, STR = BK + 8;  // +8 halfs pad
  __shared__ _Float16 As[BM * STR];
  __shared__ _Float16 Bs[BN * STR];

  const int tid = threadIdx.x, lane = tid & 31, wave = tid >> 5;
  const int wm = wave >> 1, wn = wave & 1;
  const long m0 = (long)blockIdx.y * BM;
  const long n0 = (long)blockIdx.x * BN;

  v8f acc[2][2] = {};

  for (int k0 = 0; k0 < K; k0 += BK) {
    bool tdmA = false;
#if HAS_TDM
    if (sizeof(XT) == 2) {  // f16 source: raw 2D tile -> TDM with HW padding
      if (wave == 0)
        tdm_load_tile_f16((_Float16*)As,
                          (const _Float16*)(const void*)(X + m0 * (long)K + k0),
                          BK, BM, K, /*interval 16 DW*/ 3, /*pad 4 DW*/ 3);
      tdmA = true;
    }
#endif
    if (!tdmA) {
      // Stage A tile (BM x BK) -> f16, row-major.
#pragma unroll 2
      for (int c = tid; c < BM * (BK / 8); c += 256) {
        const int row = c >> 2, coff = (c & 3) * 8;
        *(v8h*)(As + row * STR + coff) = load8(X + (m0 + row) * (long)K + k0 + coff);
      }
    }
    // Stage B tile (BN x BK) -> f16, [n][k] row-major (W row n = col n of W^T).
    for (int c = tid; c < BN * (BK / 8); c += 256) {
      const int row = c >> 2, coff = (c & 3) * 8;
      *(v8h*)(Bs + row * STR + coff) = load8(W + (n0 + row) * (long)K + k0 + coff);
    }
    if (k0 + BK < K) {  // prefetch next K-slab (global_prefetch_b8)
      __builtin_prefetch((const void*)(X + (m0 + (tid >> 1)) * (long)K + k0 + BK), 0, 1);
      __builtin_prefetch((const void*)(W + (n0 + (tid >> 2)) * (long)K + k0 + BK), 0, 1);
    }
#if HAS_TDM
    if (tdmA && wave == 0) __builtin_amdgcn_s_wait_tensorcnt(0);
#endif
    __syncthreads();

    v16h af0 = load_frag_a(As + (wm * 32 +  0) * STR, STR, lane, 0);
    v16h af1 = load_frag_a(As + (wm * 32 + 16) * STR, STR, lane, 0);
    v16h bf0 = load_frag_b(Bs + (wn * 32 +  0) * STR, STR, lane, 0);
    v16h bf1 = load_frag_b(Bs + (wn * 32 + 16) * STR, STR, lane, 0);
    acc[0][0] = wmma32(af0, bf0, acc[0][0]);
    acc[0][1] = wmma32(af0, bf1, acc[0][1]);
    acc[1][0] = wmma32(af1, bf0, acc[1][0]);
    acc[1][1] = wmma32(af1, bf1, acc[1][1]);
    __syncthreads();
  }

  // C layout: lane holds column n = lane%16; VGPR r holds row r + 8*(lane/16).
  const int n = lane & 15, kg = lane >> 4;
#pragma unroll
  for (int j = 0; j < 2; ++j) {
    const long col = n0 + wn * 32 + j * 16 + n;
    const float bv = bias[col];
#pragma unroll
    for (int i = 0; i < 2; ++i) {
      const long rowb = m0 + wm * 32 + i * 16 + 8 * kg;
#pragma unroll
      for (int r = 0; r < 8; ++r)
        store_out(Y + (rowb + r) * (long)N + col, acc[i][j][r] + bv);
    }
  }
}

// ---------------------------------------------------------------------------
// Flash attention: Q,K,V f16 in [B,S,H*DK] layout, head h at columns h*DK..
// Block: 128 threads = 4 waves; each wave owns 16 query rows (BQ=64/block).
// Key loop in chunks of BC=64; online softmax; O accumulated in WMMA regs.
// Q and K tiles staged by the TDM (wave 0); V staged manually (transpose).
// ---------------------------------------------------------------------------
__global__ __launch_bounds__(128, 2)
void flash_attn_wmma(const _Float16* __restrict__ Qp, const _Float16* __restrict__ Kp,
                     const _Float16* __restrict__ Vp, _Float16* __restrict__ Ctx) {
  constexpr int S = 2048, H = 16, DK = 64, D = 1024, BQ = 64, BC = 64;
  constexpr int STR = DK + 8;
  constexpr float scale = 0.125f;  // 1/sqrt(64)

  const int tid = threadIdx.x, lane = tid & 31, wave = tid >> 5;
  const int bh = blockIdx.y, b = bh >> 4, h = bh & (H - 1);
  const long qb = (long)blockIdx.x * BQ;

  __shared__ _Float16 Qs[BQ * STR];
  __shared__ _Float16 Ks[BC * STR];
  __shared__ _Float16 Vt[DK * STR];       // transposed: [dim][key]
  __shared__ _Float16 Ps[4][16 * STR];    // per-wave softmax-P scratch

  const _Float16* Qg = Qp + ((long)b * S + qb) * D + h * DK;
#if HAS_TDM
  if (wave == 0) {
    tdm_load_tile_f16(Qs, Qg, DK, BQ, D, /*interval 32 DW*/ 4, /*pad 4 DW*/ 3);
    __builtin_amdgcn_s_wait_tensorcnt(0);
  }
#else
  for (int c = tid; c < BQ * (DK / 8); c += 128) {
    const int row = c >> 3, coff = (c & 7) * 8;
    *(v8h*)(Qs + row * STR + coff) = *(const v8h*)(Qg + (long)row * D + coff);
  }
#endif
  __syncthreads();

  v16h aq0 = load_frag_a(Qs + wave * 16 * STR, STR, lane, 0);
  v16h aq1 = load_frag_a(Qs + wave * 16 * STR, STR, lane, 32);

  float mrow[8], lrow[8];
  v8f oacc[4] = {};
#pragma unroll
  for (int r = 0; r < 8; ++r) { mrow[r] = -3.0e38f; lrow[r] = 0.f; }

  for (int kb = 0; kb < S; kb += BC) {
    const _Float16* Kg = Kp + ((long)b * S + kb) * D + h * DK;
    const _Float16* Vg = Vp + ((long)b * S + kb) * D + h * DK;

    // Stage K tile [key][dk] — already the B-tile layout for Q@K^T.
#if HAS_TDM
    if (wave == 0) tdm_load_tile_f16(Ks, Kg, DK, BC, D, 4, 3);
#else
    for (int c = tid; c < BC * (DK / 8); c += 128) {
      const int row = c >> 3, coff = (c & 7) * 8;
      *(v8h*)(Ks + row * STR + coff) = *(const v8h*)(Kg + (long)row * D + coff);
    }
#endif
    // Stage V transposed: Vt[dim][key] so the PV B-fragment is contiguous.
    for (int c = tid; c < BC * (DK / 8); c += 128) {
      const int key = c >> 3, d0 = (c & 7) * 8;
      v8h v = *(const v8h*)(Vg + (long)key * D + d0);
#pragma unroll
      for (int j = 0; j < 8; ++j) Vt[(d0 + j) * STR + key] = v[j];
    }
    if (kb + BC < S)
      __builtin_prefetch((const void*)(Kg + (long)BC * D + (tid >> 1) * D), 0, 1);
#if HAS_TDM
    if (wave == 0) __builtin_amdgcn_s_wait_tensorcnt(0);
#endif
    __syncthreads();

    // S_ij = Q(16x64) @ K(64x64)^T : 4 n-tiles x 2 k-steps.
    v8f sacc[4] = {};
#pragma unroll
    for (int nt = 0; nt < 4; ++nt) {
      v16h b0 = load_frag_b(Ks + nt * 16 * STR, STR, lane, 0);
      sacc[nt] = wmma32(aq0, b0, sacc[nt]);
      v16h b1 = load_frag_b(Ks + nt * 16 * STR, STR, lane, 32);
      sacc[nt] = wmma32(aq1, b1, sacc[nt]);
    }

    // Online softmax (rows live across the 16 lanes of each half-wave).
    float pv[4][8];
#pragma unroll
    for (int r = 0; r < 8; ++r) {
      float mx = -3.0e38f;
#pragma unroll
      for (int nt = 0; nt < 4; ++nt) mx = fmaxf(mx, sacc[nt][r] * scale);
      mx = fmaxf(mx, __shfl_xor(mx, 1, 32));
      mx = fmaxf(mx, __shfl_xor(mx, 2, 32));
      mx = fmaxf(mx, __shfl_xor(mx, 4, 32));
      mx = fmaxf(mx, __shfl_xor(mx, 8, 32));
      const float mn = fmaxf(mrow[r], mx);
      const float alpha = __expf(mrow[r] - mn);
      float rs = 0.f;
#pragma unroll
      for (int nt = 0; nt < 4; ++nt) {
        const float p = __expf(sacc[nt][r] * scale - mn);
        pv[nt][r] = p;
        rs += p;
      }
      rs += __shfl_xor(rs, 1, 32);
      rs += __shfl_xor(rs, 2, 32);
      rs += __shfl_xor(rs, 4, 32);
      rs += __shfl_xor(rs, 8, 32);
      lrow[r] = lrow[r] * alpha + rs;
      mrow[r] = mn;
#pragma unroll
      for (int dt = 0; dt < 4; ++dt) oacc[dt][r] *= alpha;
    }

    // C-layout -> A-fragment layout via per-wave LDS round trip (LDS ops from
    // the same wave are in order, no barrier needed).
    {
      const int n = lane & 15, kg = lane >> 4;
      _Float16* pw = Ps[wave];
#pragma unroll
      for (int nt = 0; nt < 4; ++nt)
#pragma unroll
        for (int r = 0; r < 8; ++r)
          pw[(r + 8 * kg) * STR + nt * 16 + n] = (_Float16)pv[nt][r];
    }

    // O += P(16x64) @ V(64x64) : contraction over keys, 2 k-steps x 4 d-tiles.
#pragma unroll
    for (int ks = 0; ks < 2; ++ks) {
      v16h ap = load_frag_a(Ps[wave], STR, lane, ks * 32);
#pragma unroll
      for (int dt = 0; dt < 4; ++dt) {
        v16h bv = load_frag_b(Vt + dt * 16 * STR, STR, lane, ks * 32);
        oacc[dt] = wmma32(ap, bv, oacc[dt]);
      }
    }
    __syncthreads();
  }

  // Normalize and scatter to ctx [B,S,H*DK] (same layout as Q/K/V).
  const int n = lane & 15, kg = lane >> 4;
#pragma unroll
  for (int r = 0; r < 8; ++r) {
    const float inv = 1.f / lrow[r];
    const long srow = qb + wave * 16 + 8 * kg + r;
    _Float16* cg = Ctx + ((long)b * S + srow) * D + h * DK;
#pragma unroll
    for (int dt = 0; dt < 4; ++dt)
      cg[dt * 16 + n] = (_Float16)(oacc[dt][r] * inv);
  }
}

// ---------------------------------------------------------------------------
extern "C" void kernel_launch(void* const* d_in, const int* in_sizes, int n_in,
                              void* d_out, int out_size, void* d_ws, size_t ws_size,
                              hipStream_t stream) {
  constexpr int Bb = 4, Ss = 2048, Dd = 1024;
  constexpr int M = Bb * Ss;                 // 8192
  constexpr long NE = (long)Bb * Ss * Dd;    // 8,388,608 elems per f16 tensor

  const float* q  = (const float*)d_in[0];
  const float* k  = (const float*)d_in[1];
  const float* v  = (const float*)d_in[2];
  const float* Wq = (const float*)d_in[3];
  const float* bq = (const float*)d_in[4];
  const float* Wk = (const float*)d_in[5];
  const float* bk = (const float*)d_in[6];
  const float* Wv = (const float*)d_in[7];
  const float* bv = (const float*)d_in[8];
  const float* Wo = (const float*)d_in[9];
  const float* bo = (const float*)d_in[10];

  // Workspace: Qp | Kp | Vp | Ctx (f16), 4 * 16MB = 64MB required.
  _Float16* Qp = (_Float16*)d_ws;
  _Float16* Kp = Qp + NE;
  _Float16* Vp = Kp + NE;
  _Float16* Cx = Vp + NE;

  dim3 gb(Dd / 64, M / 128);  // (16, 64)
  gemm_bias_wmma<float, _Float16><<<gb, 256, 0, stream>>>(q, Wq, bq, Qp, M, Dd, Dd);
  gemm_bias_wmma<float, _Float16><<<gb, 256, 0, stream>>>(k, Wk, bk, Kp, M, Dd, Dd);
  gemm_bias_wmma<float, _Float16><<<gb, 256, 0, stream>>>(v, Wv, bv, Vp, M, Dd, Dd);

  flash_attn_wmma<<<dim3(Ss / 64, Bb * 16), 128, 0, stream>>>(Qp, Kp, Vp, Cx);

  gemm_bias_wmma<_Float16, float><<<gb, 256, 0, stream>>>(Cx, Wo, bo, (float*)d_out,
                                                          M, Dd, Dd);
}
